// KG_extract_83459804496224
// MI455X (gfx1250) — matrix operand order repaired and checked by reading, not compiled, not run on hardware
//
#include <hip/hip_runtime.h>
#include <hip/hip_bf16.h>
#include <math.h>

// ---------------- problem constants ----------------
#define HID    1024
#define E_DIM  40000
#define R_DIM  500
#define R_PAD  512
#define B_DIM  256
#define K_CAND 128

// ---------------- workspace layout (bytes) ----------------
#define OFF_ACC   0                                    // 16 floats of accumulators
#define OFF_XB    4096                                 // 256*1024 bf16
#define OFF_HLOG  (OFF_XB   + (size_t)B_DIM*HID*2)     // 256*40000 f32
#define OFF_TLOG  (OFF_HLOG + (size_t)B_DIM*E_DIM*4)   // 256*40000 f32
#define OFF_RLOG  (OFF_TLOG + (size_t)B_DIM*E_DIM*4)   // 256*512 f32 (padded)
#define OFF_XREC  (OFF_RLOG + (size_t)B_DIM*R_PAD*4)   // 256*1024 f32
#define OFF_HACT  (OFF_XREC + (size_t)B_DIM*HID*4)     // 256*40000 bf16
#define OFF_TACT  (OFF_HACT + (size_t)B_DIM*E_DIM*2)   // 256*40000 bf16
#define OFF_RACT  (OFF_TACT + (size_t)B_DIM*E_DIM*2)   // 256*512 bf16 (padded)
#define OFF_WER   (OFF_RACT + (size_t)B_DIM*R_PAD*2)   // 1024*512 f32 (padded We_r)
#define OFF_WDR   (OFF_WER  + (size_t)HID*R_PAD*4)     // 512*1024 f32 (padded Wd_r)

// acc slots: 0 sumsq, 1 sp_h, 2 sp_r, 3 sp_t, 4 tgt_h, 5 tgt_r, 6 tgt_t,
//            7 sum_ce, 8 sum_valid, 9 inv_t

typedef __attribute__((ext_vector_type(16))) __bf16 v16bf;
typedef __attribute__((ext_vector_type(8)))  float  v8f;

union ABFrag { v16bf v; unsigned short e[16]; unsigned int u32[8]; uint4 u4[2]; };
union CFrag  { v8f   v; float f[8]; };

// scalar f32 -> bf16, round-to-nearest-even
__device__ __forceinline__ unsigned short f2bf(float f) {
  unsigned int u = __float_as_uint(f);
  unsigned int r = u + 0x7FFFu + ((u >> 16) & 1u);
  return (unsigned short)(r >> 16);
}

// pack two f32 -> packed bf16x2 with round-half-up; one v_perm_b32 grabs both high halves
__device__ __forceinline__ unsigned int pack2bf(float lo, float hi) {
  unsigned int ulo = __float_as_uint(lo) + 0x8000u;
  unsigned int uhi = __float_as_uint(hi) + 0x8000u;
  // dst = { uhi[31:16], ulo[31:16] } : sel bytes {7,6,3,2} over {S0=uhi, S1=ulo}
  return __builtin_amdgcn_perm(uhi, ulo, 0x07060302u);
}

__device__ __forceinline__ float softplus_f(float x) {
  return fmaxf(x, 0.f) + log1pf(expf(-fabsf(x)));
}

__device__ __forceinline__ void block_reduce_atomic(float v, float* target) {
  __shared__ float sm[256];
  sm[threadIdx.x] = v;
  __syncthreads();
  for (int s = blockDim.x >> 1; s > 0; s >>= 1) {
    if ((int)threadIdx.x < s) sm[threadIdx.x] += sm[threadIdx.x + s];
    __syncthreads();
  }
  if (threadIdx.x == 0) atomicAdd(target, sm[0]);
}

// ---------------- setup kernels ----------------
__global__ void kg_init_acc(const float* inv_t_param, float* acc) {
  if (threadIdx.x == 0) {
    for (int i = 0; i < 16; ++i) acc[i] = 0.f;
    acc[9] = fminf(softplus_f(inv_t_param[0]), 100.f);
  }
}

__global__ void kg_conv_x(const float* __restrict__ x, unsigned short* __restrict__ xb) {
  int i = blockIdx.x * blockDim.x + threadIdx.x;
  if (i < B_DIM * HID) xb[i] = f2bf(x[i]);
}

__global__ void kg_init_xrec(const float* bd_h, const float* bd_r, const float* bd_t,
                             float* __restrict__ xrec) {
  int i = blockIdx.x * blockDim.x + threadIdx.x;
  if (i < B_DIM * HID) {
    int c = i & (HID - 1);
    xrec[i] = bd_h[c] + bd_r[c] + bd_t[c];
  }
}

__global__ void kg_pad_wer(const float* __restrict__ Wer, float* __restrict__ out) {
  int i = blockIdx.x * blockDim.x + threadIdx.x;       // 1024*512
  if (i < HID * R_PAD) {
    int k = i >> 9, n = i & (R_PAD - 1);
    out[i] = (n < R_DIM) ? Wer[k * R_DIM + n] : 0.f;
  }
}

__global__ void kg_pad_wdr(const float* __restrict__ Wdr, float* __restrict__ out) {
  int i = blockIdx.x * blockDim.x + threadIdx.x;       // 512*1024
  if (i < R_PAD * HID) {
    int k = i >> 10;
    out[i] = (k < R_DIM) ? Wdr[i] : 0.f;
  }
}

// ---------------- encoder GEMM ----------------
// One wave owns a 16-column N-strip x ALL 256 rows (16 accumulators = 128 VGPRs).
// Weights stream from HBM exactly once; xb (512 KB bf16) stays L2-resident.
// N/LDOUT compile-time => strided B loads become one clause with immediate offsets.
template<int N, int NREAL, int LDOUT>
__global__ void __launch_bounds__(128)
kg_enc_gemm(const unsigned short* __restrict__ xb, const float* __restrict__ W,
            const float* __restrict__ bias, float* __restrict__ logits,
            unsigned short* __restrict__ acts) {
  const int wave   = threadIdx.x >> 5;
  const int lane   = threadIdx.x & 31;
  const int n0     = (blockIdx.x * 4 + wave) * 16;
  const int lanelo = lane & 15;
  const int hi     = lane >> 4;
  const int khalf  = hi * 16;   // A: K half select per lane half
  const int koff   = hi * 16;   // B: K half select per lane half
  const int nw     = n0 + lanelo;

  CFrag acc[16];
#pragma unroll
  for (int m = 0; m < 16; ++m)
#pragma unroll
    for (int j = 0; j < 8; ++j) acc[m].f[j] = 0.f;

  const float* wp = W + (size_t)koff * N + nw;     // per-lane column base; +32*N per kstep
  const unsigned short* ap = xb + (size_t)lanelo * HID;

  for (int k0 = 0; k0 < HID; k0 += 32) {
    __builtin_prefetch(wp + 32 * (size_t)N, 0, 1); // global_prefetch_b8 ahead of the stream
    ABFrag b;
#pragma unroll
    for (int i = 0; i < 8; ++i)                    // 16 coalesced b32 loads, const offsets
      b.u32[i] = pack2bf(wp[(size_t)(2 * i) * N], wp[(size_t)(2 * i + 1) * N]);
#pragma unroll
    for (int m = 0; m < 16; ++m) {                 // reuse B across all 16 M-tiles
      ABFrag a;
      const unsigned short* ar = ap + (size_t)m * 16 * HID + k0 + khalf;
      a.u4[0] = *(const uint4*)(ar);
      a.u4[1] = *(const uint4*)(ar + 8);
      acc[m].v = __builtin_amdgcn_wmma_f32_16x16x32_bf16(
          false, a.v, false, b.v, (short)0, acc[m].v, false, false);
    }
    wp += 32 * (size_t)N;
  }

  // epilogue: logits f32 + relu->bf16 acts.  C/D layout: VGPR j -> row m*16+j(+8), col nw
  const bool nvalid = (nw < NREAL);
  const float bval = nvalid ? bias[nw] : 0.f;
#pragma unroll
  for (int m = 0; m < 16; ++m) {
    const int rbase = m * 16 + hi * 8;
#pragma unroll
    for (int j = 0; j < 8; ++j) {
      const float val = nvalid ? (acc[m].f[j] + bval) : 0.f;
      const size_t idx = (size_t)(rbase + j) * LDOUT + nw;
      logits[idx] = val;
      acts[idx] = f2bf(fmaxf(val, 0.f));
    }
  }
}

// ---------------- decoder GEMM: xrec += acts(256xK,bf16) @ Wd(Kx1024) ----------------
// Same all-M blocking; split-K over grid.z with f32 global atomics into xrec.
template<int LDACT>
__global__ void __launch_bounds__(128)
kg_dec_gemm(const unsigned short* __restrict__ acts, const float* __restrict__ Wd,
            float* __restrict__ xrec, int ksteps) {
  const int wave   = threadIdx.x >> 5;
  const int lane   = threadIdx.x & 31;
  const int n0     = (blockIdx.x * 4 + wave) * 16;   // 64 strips over N=1024
  const int lanelo = lane & 15;
  const int hi     = lane >> 4;
  const int khalf  = hi * 16;
  const int koff   = hi * 16;
  const int nw     = n0 + lanelo;
  const int kstart = blockIdx.z * ksteps * 32;

  CFrag acc[16];
#pragma unroll
  for (int m = 0; m < 16; ++m)
#pragma unroll
    for (int j = 0; j < 8; ++j) acc[m].f[j] = 0.f;

  const float* wp = Wd + (size_t)(kstart + koff) * HID + nw;
  const unsigned short* ap = acts + (size_t)lanelo * LDACT + kstart + khalf;

  for (int s = 0; s < ksteps; ++s) {
    __builtin_prefetch(wp + 32 * HID, 0, 1);
    ABFrag b;
#pragma unroll
    for (int i = 0; i < 8; ++i)                      // stride HID -> small const offsets
      b.u32[i] = pack2bf(wp[(2 * i) * HID], wp[(2 * i + 1) * HID]);
#pragma unroll
    for (int m = 0; m < 16; ++m) {
      ABFrag a;
      const unsigned short* ar = ap + (size_t)m * 16 * LDACT;
      a.u4[0] = *(const uint4*)(ar);
      a.u4[1] = *(const uint4*)(ar + 8);
      acc[m].v = __builtin_amdgcn_wmma_f32_16x16x32_bf16(
          false, a.v, false, b.v, (short)0, acc[m].v, false, false);
    }
    wp += 32 * HID;
    ap += 32;
  }

#pragma unroll
  for (int m = 0; m < 16; ++m) {
    const int rbase = m * 16 + hi * 8;
#pragma unroll
    for (int j = 0; j < 8; ++j)
      atomicAdd(&xrec[(size_t)(rbase + j) * HID + nw], acc[m].f[j]);
  }
}

// ---------------- reductions ----------------
__global__ void kg_softplus_sum(const float* __restrict__ logits, int cols, int ld,
                                float* acc_slot) {
  const size_t total = (size_t)B_DIM * cols;
  float s = 0.f;
  for (size_t i = blockIdx.x * (size_t)blockDim.x + threadIdx.x; i < total;
       i += (size_t)gridDim.x * blockDim.x) {
    const size_t r = i / cols, cc = i % cols;
    s += softplus_f(logits[r * ld + cc]);
  }
  block_reduce_atomic(s, acc_slot);
}

__global__ void kg_recon_sum(const float* __restrict__ xrec, const float* __restrict__ x,
                             float* acc_slot) {
  float s = 0.f;
  for (size_t i = blockIdx.x * (size_t)blockDim.x + threadIdx.x; i < (size_t)B_DIM * HID;
       i += (size_t)gridDim.x * blockDim.x) {
    const float d = xrec[i] - x[i];
    s += d * d;
  }
  block_reduce_atomic(s, acc_slot);
}

// ---------------- sparse BCE target terms ----------------
__global__ void kg_target_gather(const float* __restrict__ hlog, const float* __restrict__ rlog,
                                 const float* __restrict__ tlog, const int* __restrict__ eid,
                                 const int* __restrict__ trip, const unsigned char* __restrict__ istail,
                                 float* acc) {
  const int b = threadIdx.x;   // one block of 256
  const bool tail = istail[b] != 0;
  const int t0 = trip[b * 3 + 0], t1 = trip[b * 3 + 1], t2 = trip[b * 3 + 2];

  const float tr = rlog[(size_t)b * R_PAD + t1];
  float cand_h = 0.f, cand_t = 0.f;
  for (int k = 0; k < K_CAND; ++k) {
    const int id = eid[b * K_CAND + k];
    bool dup = false;
    for (int j = 0; j < k; ++j)
      if (eid[b * K_CAND + j] == id) { dup = true; break; }
    if (!dup) {   // .set(1.0) scatter semantics: duplicates count once
      cand_h += hlog[(size_t)b * E_DIM + id];
      cand_t += tlog[(size_t)b * E_DIM + id];
    }
  }
  float th, tt;
  if (tail) { th = hlog[(size_t)b * E_DIM + t0]; tt = cand_t; }
  else      { th = cand_h;                       tt = tlog[(size_t)b * E_DIM + t2]; }
  atomicAdd(&acc[4], th);
  atomicAdd(&acc[5], tr);
  atomicAdd(&acc[6], tt);
}

// ---------------- KGC loss: one block (4 waves) per batch row ----------------
__global__ void __launch_bounds__(128)
kg_kgc(const float* __restrict__ Wd_h, const float* __restrict__ Wd_r,
       const float* __restrict__ Wd_t, const int* __restrict__ eid,
       const int* __restrict__ trip, const unsigned char* __restrict__ istail,
       float* acc) {
  __shared__ float base[HID];
  __shared__ float dist[K_CAND];
  const int b = blockIdx.x;
  const bool tail = istail[b] != 0;
  const int t0 = trip[b * 3 + 0], t1 = trip[b * 3 + 1], t2 = trip[b * 3 + 2];
  const float* A  = tail ? (Wd_h + (size_t)t0 * HID) : (Wd_t + (size_t)t2 * HID);
  const float* Rw = Wd_r + (size_t)t1 * HID;
  for (int i = threadIdx.x; i < HID; i += blockDim.x)
    base[i] = tail ? (A[i] + Rw[i]) : (A[i] - Rw[i]);   // ||base - cand|| either way
  __syncthreads();

  const int wave = threadIdx.x >> 5, lane = threadIdx.x & 31;
  const float* Wc = tail ? Wd_t : Wd_h;
  for (int k = wave; k < K_CAND; k += 4) {
    const float* cand = Wc + (size_t)eid[b * K_CAND + k] * HID;
    float p = 0.f;
    for (int i = lane; i < HID; i += 32) {
      const float d = base[i] - cand[i];
      p += d * d;
    }
    for (int off = 16; off > 0; off >>= 1) p += __shfl_down(p, off, 32);
    if (lane == 0) dist[k] = sqrtf(p);
  }
  __syncthreads();

  if (threadIdx.x == 0) {
    const float inv_t = acc[9];
    const int target = tail ? t2 : t0;
    float mx = -1e30f; int label = -1;
    for (int k = 0; k < K_CAND; ++k) {
      const float lg = -dist[k] * inv_t;
      dist[k] = lg;
      if (lg > mx) mx = lg;
      if (label < 0 && eid[b * K_CAND + k] == target) label = k;
    }
    if (label >= 0) {
      float se = 0.f;
      for (int k = 0; k < K_CAND; ++k) se += expf(dist[k] - mx);
      const float ce = (logf(se) + mx) - dist[label];
      atomicAdd(&acc[7], ce);
      atomicAdd(&acc[8], 1.f);
    }
  }
}

__global__ void kg_finalize(const float* acc, float* out) {
  if (threadIdx.x == 0) {
    const float recon = acc[0] / (float)(B_DIM * HID);
    const float lbl = (acc[1] - acc[4]) / (float)(B_DIM * E_DIM)
                    + (acc[2] - acc[5]) / (float)(B_DIM * R_DIM)
                    + (acc[3] - acc[6]) / (float)(B_DIM * E_DIM);
    const float kgc = acc[7] / (acc[8] + 1e-8f);
    out[0] = recon + lbl + kgc;
  }
}

// ---------------- host launcher ----------------
extern "C" void kernel_launch(void* const* d_in, const int* in_sizes, int n_in,
                              void* d_out, int out_size, void* d_ws, size_t ws_size,
                              hipStream_t stream) {
  const float* x     = (const float*)d_in[0];
  const float* We_h  = (const float*)d_in[1];
  const float* be_h  = (const float*)d_in[2];
  const float* We_r  = (const float*)d_in[3];
  const float* be_r  = (const float*)d_in[4];
  const float* We_t  = (const float*)d_in[5];
  const float* be_t  = (const float*)d_in[6];
  const float* Wd_h  = (const float*)d_in[7];
  const float* bd_h  = (const float*)d_in[8];
  const float* Wd_r  = (const float*)d_in[9];
  const float* bd_r  = (const float*)d_in[10];
  const float* Wd_t  = (const float*)d_in[11];
  const float* bd_t  = (const float*)d_in[12];
  const float* inv_t = (const float*)d_in[13];
  const int* entity  = (const int*)d_in[14];
  const int* triple  = (const int*)d_in[15];
  const unsigned char* istail = (const unsigned char*)d_in[16];

  char* ws = (char*)d_ws;
  float*          acc   = (float*)(ws + OFF_ACC);
  unsigned short* xb    = (unsigned short*)(ws + OFF_XB);
  float*          hlog  = (float*)(ws + OFF_HLOG);
  float*          tlog  = (float*)(ws + OFF_TLOG);
  float*          rlog  = (float*)(ws + OFF_RLOG);
  float*          xrec  = (float*)(ws + OFF_XREC);
  unsigned short* hact  = (unsigned short*)(ws + OFF_HACT);
  unsigned short* tact  = (unsigned short*)(ws + OFF_TACT);
  unsigned short* ract  = (unsigned short*)(ws + OFF_RACT);
  float*          werp  = (float*)(ws + OFF_WER);
  float*          wdrp  = (float*)(ws + OFF_WDR);
  float*          out   = (float*)d_out;

  kg_init_acc<<<1, 32, 0, stream>>>(inv_t, acc);
  kg_conv_x<<<(B_DIM * HID + 255) / 256, 256, 0, stream>>>(x, xb);
  kg_init_xrec<<<(B_DIM * HID + 255) / 256, 256, 0, stream>>>(bd_h, bd_r, bd_t, xrec);
  kg_pad_wer<<<(HID * R_PAD + 255) / 256, 256, 0, stream>>>(We_r, werp);
  kg_pad_wdr<<<(R_PAD * HID + 255) / 256, 256, 0, stream>>>(Wd_r, wdrp);

  // encoders: grid.x = N-strips/4 (4 waves per block, each wave = 16 cols x all 256 rows)
  kg_enc_gemm<E_DIM, E_DIM, E_DIM><<<E_DIM / 64, 128, 0, stream>>>(xb, We_h, be_h, hlog, hact);
  kg_enc_gemm<E_DIM, E_DIM, E_DIM><<<E_DIM / 64, 128, 0, stream>>>(xb, We_t, be_t, tlog, tact);
  kg_enc_gemm<R_PAD, R_DIM, R_PAD><<<R_PAD / 64, 128, 0, stream>>>(xb, werp, be_r, rlog, ract);

  kg_softplus_sum<<<2048, 256, 0, stream>>>(hlog, E_DIM, E_DIM, acc + 1);
  kg_softplus_sum<<<128,  256, 0, stream>>>(rlog, R_DIM, R_PAD, acc + 2);
  kg_softplus_sum<<<2048, 256, 0, stream>>>(tlog, E_DIM, E_DIM, acc + 3);

  // decoders: 16 blocks x 4 waves = 64 N-strips; split-K over grid.z
  kg_dec_gemm<E_DIM><<<dim3(16, 1, 10), 128, 0, stream>>>(hact, Wd_h, xrec, 125);
  kg_dec_gemm<E_DIM><<<dim3(16, 1, 10), 128, 0, stream>>>(tact, Wd_t, xrec, 125);
  kg_dec_gemm<R_PAD><<<dim3(16, 1, 1),  128, 0, stream>>>(ract, wdrp, xrec, 16);

  kg_recon_sum<<<512, 256, 0, stream>>>(xrec, x, acc + 0);
  kg_target_gather<<<1, 256, 0, stream>>>(hlog, rlog, tlog, entity, triple, istail, acc);
  kg_kgc<<<B_DIM, 128, 0, stream>>>(Wd_h, Wd_r, Wd_t, entity, triple, istail, acc);
  kg_finalize<<<1, 32, 0, stream>>>(acc, out);
}
